// GraphSAGE_13520557047869
// MI455X (gfx1250) — matrix operand order repaired and checked by reading, not compiled, not run on hardware
//
#include <hip/hip_runtime.h>

#define NN 4096
#define DF 128
#define BATCH 4
#define KSPLIT 4
#define KLEN (NN / KSPLIT)

typedef __attribute__((ext_vector_type(16))) _Float16 v16h;
typedef __attribute__((ext_vector_type(8)))  _Float16 h8;
typedef __attribute__((ext_vector_type(4)))  _Float16 h4;
typedef __attribute__((ext_vector_type(8)))  float    v8f;

// Build a 16-half WMMA fragment from two 16-byte-aligned groups of 8 halfs.
__device__ __forceinline__ v16h make_frag(const _Float16* lo, const _Float16* hi) {
  h8 a = *(const h8*)lo;
  h8 b = *(const h8*)hi;
  v16h r;
#pragma unroll
  for (int e = 0; e < 8; ++e) { r[e] = a[e]; r[e + 8] = b[e]; }
  return r;
}

// ---------------------------------------------------------------------------
// Degree: deg[b,i] = sum_j (adj[b,j,i] != 0).  Split over j-chunks + atomics.
// ---------------------------------------------------------------------------
__global__ __launch_bounds__(256) void deg_partial_kernel(const int* __restrict__ adj,
                                                          float* __restrict__ degbuf) {
  const int b  = blockIdx.z;
  const int i  = blockIdx.x * 256 + threadIdx.x;
  const int j0 = blockIdx.y * 256;
  const int* base = adj + (size_t)b * NN * NN + i;
  int cnt = 0;
#pragma unroll 4
  for (int j = 0; j < 256; ++j)
    cnt += (base[(size_t)(j0 + j) * NN] != 0);
  atomicAdd(&degbuf[b * NN + i], (float)cnt);
}

__global__ void invdeg_kernel(const float* __restrict__ degbuf, float* __restrict__ invdeg) {
  int idx = blockIdx.x * blockDim.x + threadIdx.x;
  if (idx < BATCH * NN) invdeg[idx] = 1.0f / fmaxf(degbuf[idx], 1.0f);
}

// features f32 -> f16, 8 elements per thread
__global__ void cvt_kernel(const float* __restrict__ src, _Float16* __restrict__ dst, int n8) {
  int idx = blockIdx.x * blockDim.x + threadIdx.x;
  if (idx >= n8) return;
  const float4* s = (const float4*)src + 2 * (size_t)idx;
  float4 a = s[0], b = s[1];
  h8 o;
  o[0] = (_Float16)a.x; o[1] = (_Float16)a.y; o[2] = (_Float16)a.z; o[3] = (_Float16)a.w;
  o[4] = (_Float16)b.x; o[5] = (_Float16)b.y; o[6] = (_Float16)b.z; o[7] = (_Float16)b.w;
  ((h8*)dst)[idx] = o;
}

// ---------------------------------------------------------------------------
// Partial aggregate: AGGP[split][b,i,:] = sum_{j in split} A[b,j,i] * X[b,j,:]
// WMMA f16->f32.  Block: 256 threads (8 waves), tile 128 nodes x 128 features,
// K-range = 1024 per split (4-way K-split => 512 blocks => HBM saturation).
// ---------------------------------------------------------------------------
__global__ __launch_bounds__(256, 1) void agg_kernel(const int* __restrict__ adj,
                                                     const _Float16* __restrict__ X,
                                                     float* __restrict__ AGGP) {
  constexpr int STR = 40;                   // half-stride: 80B rows (16B aligned, bank-clean)
  __shared__ ushort   ldsAT[128 * STR];     // A^T tile: [i][k], raw f16 bits 0x3C00 / 0
  __shared__ _Float16 ldsXT[128 * STR];     // X^T tile: [d][k]

  const int b     = blockIdx.z;
  const int split = blockIdx.y;
  const int i0    = blockIdx.x * 128;
  const int kbase = split * KLEN;
  const int tid   = threadIdx.x;
  const int wave  = tid >> 5;
  const int lane  = tid & 31;

  const int*      adjB = adj + (size_t)b * NN * NN;
  const _Float16* Xb   = X   + (size_t)b * NN * DF;

  v8f acc[8];
#pragma unroll
  for (int g = 0; g < 8; ++g)
#pragma unroll
    for (int r = 0; r < 8; ++r) acc[g][r] = 0.0f;

  const int kload = tid >> 3;          // 0..31  (k within tile)
  const int iload = (tid & 7) * 16;    // 0..112 (i/d within tile)

  const int arow = 16 * wave + (lane & 15);
  const int koff = (lane >> 4) * 8;    // A frag: halfs[0..7]=K koff.., halfs[8..15]=K koff+16..
  const int kb   = (lane >> 4) * 16;   // unused; B frag takes the same koff K split as A
  const int lcol = lane & 15;

  const int*      adjRow = adjB + (size_t)kload * NN + i0 + iload;
  const _Float16* xRow   = Xb   + (size_t)kload * DF + iload;

  // prologue: fetch first tile of this split into registers
  int4 pa[4];
  h8   px[2];
  {
    const int4* s = (const int4*)(adjRow + (size_t)kbase * NN);
    pa[0] = s[0]; pa[1] = s[1]; pa[2] = s[2]; pa[3] = s[3];
    const h8* t = (const h8*)(xRow + (size_t)kbase * DF);
    px[0] = t[0]; px[1] = t[1];
  }

  for (int k0 = kbase; k0 < kbase + KLEN; k0 += 32) {
    __syncthreads();
    // stage current tile (registers -> LDS, transposed)
#pragma unroll
    for (int v = 0; v < 4; ++v) {
      ldsAT[(iload + 4 * v + 0) * STR + kload] = pa[v].x != 0 ? (ushort)0x3C00 : (ushort)0;
      ldsAT[(iload + 4 * v + 1) * STR + kload] = pa[v].y != 0 ? (ushort)0x3C00 : (ushort)0;
      ldsAT[(iload + 4 * v + 2) * STR + kload] = pa[v].z != 0 ? (ushort)0x3C00 : (ushort)0;
      ldsAT[(iload + 4 * v + 3) * STR + kload] = pa[v].w != 0 ? (ushort)0x3C00 : (ushort)0;
    }
#pragma unroll
    for (int v = 0; v < 2; ++v)
#pragma unroll
      for (int e = 0; e < 8; ++e)
        ldsXT[(iload + 8 * v + e) * STR + kload] = px[v][e];
    __syncthreads();

    // Issue ALL fragment loads first (A then 8 distinct B octets) ...
    v16h afrag = make_frag((const _Float16*)&ldsAT[arow * STR + koff],
                           (const _Float16*)&ldsAT[arow * STR + koff + 16]);
    v16h bf[8];
#pragma unroll
    for (int g = 0; g < 8; ++g) {
      const int bcol = 16 * g + lcol;
      bf[g] = make_frag(&ldsXT[bcol * STR + koff], &ldsXT[bcol * STR + koff + 16]);
    }

    // ... then overlap the next tile's global fetch with the in-flight LDS loads
    if (k0 + 32 < kbase + KLEN) {
      const int4* s = (const int4*)(adjRow + (size_t)(k0 + 32) * NN);
      pa[0] = s[0]; pa[1] = s[1]; pa[2] = s[2]; pa[3] = s[3];
      const h8* t = (const h8*)(xRow + (size_t)(k0 + 32) * DF);
      px[0] = t[0]; px[1] = t[1];
    }
    if (k0 + 96 < NN)   // stream hint three tiles ahead on the adjacency
      __builtin_prefetch(adjRow + (size_t)(k0 + 96) * NN, 0, 1);

    // ... then the WMMA burst (graduated dscnt waits, back-to-back XDL issue)
#pragma unroll
    for (int g = 0; g < 8; ++g)
      acc[g] = __builtin_amdgcn_wmma_f32_16x16x32_f16(
          false, afrag, false, bf[g], (short)0, acc[g], false, false);
  }

  // Epilogue: store f32 partials. C layout: VGPR r -> row r (lanes 0-15) / r+8 (16-31).
  const int iTile = i0 + 16 * wave;
  const int lhalf = lane >> 4;
  float* outP = AGGP + ((size_t)split * BATCH + b) * NN * DF;
#pragma unroll
  for (int g = 0; g < 8; ++g)
#pragma unroll
    for (int r = 0; r < 8; ++r)
      outP[(size_t)(iTile + r + 8 * lhalf) * DF + 16 * g + lcol] = acc[g][r];
}

// ---------------------------------------------------------------------------
// AGG[b,i,d] = (sum_s AGGP[s][b,i,d]) * invdeg[b,i]  -> f16 (deterministic order)
// ---------------------------------------------------------------------------
__global__ void reduce_kernel(const float* __restrict__ P, const float* __restrict__ invdeg,
                              _Float16* __restrict__ AGG) {
  int idx = blockIdx.x * blockDim.x + threadIdx.x;          // quad index
  const int total4 = BATCH * NN * DF / 4;
  if (idx >= total4) return;
  const size_t base   = (size_t)idx * 4;
  const size_t stride = (size_t)BATCH * NN * DF;
  float4 s0 = *(const float4*)(P + base);
  float4 s1 = *(const float4*)(P + stride + base);
  float4 s2 = *(const float4*)(P + 2 * stride + base);
  float4 s3 = *(const float4*)(P + 3 * stride + base);
  float inv = invdeg[base / DF];
  h4 o;
  o[0] = (_Float16)(((s0.x + s1.x) + (s2.x + s3.x)) * inv);
  o[1] = (_Float16)(((s0.y + s1.y) + (s2.y + s3.y)) * inv);
  o[2] = (_Float16)(((s0.z + s1.z) + (s2.z + s3.z)) * inv);
  o[3] = (_Float16)(((s0.w + s1.w) + (s2.w + s3.w)) * inv);
  *(h4*)(AGG + base) = o;
}

// ---------------------------------------------------------------------------
// H = sigmoid([X | AGG] @ W + bias);  Xout = H / (||H||_2 + eps)   (f16 out)
// K = 256 (first 128 from X, second 128 from AGG).
// ---------------------------------------------------------------------------
__global__ __launch_bounds__(256, 1) void dense_kernel(const _Float16* __restrict__ Xin,
                                                       const _Float16* __restrict__ AGGin,
                                                       const float* __restrict__ W,
                                                       const float* __restrict__ bias,
                                                       _Float16* __restrict__ Xout) {
  constexpr int STR = 40;
  __shared__ _Float16 ldsWT[128 * STR];   // W^T tile: [o][k] f16

  const int b    = blockIdx.y;
  const int i0   = blockIdx.x * 128;
  const int tid  = threadIdx.x;
  const int wave = tid >> 5;
  const int lane = tid & 31;

  const _Float16* Xb = Xin   + (size_t)b * NN * DF;
  const _Float16* Ab = AGGin + (size_t)b * NN * DF;

  v8f acc[8];
#pragma unroll
  for (int g = 0; g < 8; ++g)
#pragma unroll
    for (int r = 0; r < 8; ++r) acc[g][r] = 0.0f;

  const int kload = tid >> 3;
  const int oload = (tid & 7) * 16;

  const int arow = i0 + 16 * wave + (lane & 15);
  const int koff = (lane >> 4) * 8;
  const int kb   = (lane >> 4) * 16;
  const int lcol = lane & 15;

  const float* wRow = W + (size_t)kload * DF + oload;

  float4 pw[4];
  {
    const float4* s = (const float4*)wRow;
    pw[0] = s[0]; pw[1] = s[1]; pw[2] = s[2]; pw[3] = s[3];
  }

  for (int k0 = 0; k0 < 2 * DF; k0 += 32) {
    __syncthreads();
#pragma unroll
    for (int v = 0; v < 4; ++v) {
      ldsWT[(oload + 4 * v + 0) * STR + kload] = (_Float16)pw[v].x;
      ldsWT[(oload + 4 * v + 1) * STR + kload] = (_Float16)pw[v].y;
      ldsWT[(oload + 4 * v + 2) * STR + kload] = (_Float16)pw[v].z;
      ldsWT[(oload + 4 * v + 3) * STR + kload] = (_Float16)pw[v].w;
    }
    __syncthreads();

    // A operand straight from global (rows are contiguous f16, 16B aligned)
    const _Float16* asrc = (k0 < DF) ? (Xb + (size_t)arow * DF + k0)
                                     : (Ab + (size_t)arow * DF + (k0 - DF));
    v16h afrag = make_frag(asrc + koff, asrc + koff + 16);
    v16h bf[8];
#pragma unroll
    for (int g = 0; g < 8; ++g) {
      const int bcol = 16 * g + lcol;
      bf[g] = make_frag(&ldsWT[bcol * STR + koff], &ldsWT[bcol * STR + koff + 16]);
    }

    if (k0 + 32 < 2 * DF) {
      const float4* s = (const float4*)(wRow + (size_t)(k0 + 32) * DF);
      pw[0] = s[0]; pw[1] = s[1]; pw[2] = s[2]; pw[3] = s[3];
    }

#pragma unroll
    for (int g = 0; g < 8; ++g)
      acc[g] = __builtin_amdgcn_wmma_f32_16x16x32_f16(
          false, afrag, false, bf[g], (short)0, acc[g], false, false);
  }

  // Epilogue: bias + sigmoid, row L2 norm (in-lane over 8 groups, then 16-lane butterfly)
  const int lhalf = lane >> 4;
  float h[8][8];
#pragma unroll
  for (int g = 0; g < 8; ++g) {
    float bv = bias[16 * g + lcol];
#pragma unroll
    for (int r = 0; r < 8; ++r) {
      float v = acc[g][r] + bv;
      h[g][r] = 1.0f / (1.0f + __expf(-v));
    }
  }
  float rs[8];
#pragma unroll
  for (int r = 0; r < 8; ++r) {
    float s = 0.0f;
#pragma unroll
    for (int g = 0; g < 8; ++g) s += h[g][r] * h[g][r];
#pragma unroll
    for (int m = 1; m < 16; m <<= 1) s += __shfl_xor(s, m, 32);
    rs[r] = 1.0f / (sqrtf(s) + 1e-6f);
  }
  const int iTile = i0 + 16 * wave;
  _Float16* outB = Xout + (size_t)b * NN * DF;
#pragma unroll
  for (int g = 0; g < 8; ++g)
#pragma unroll
    for (int r = 0; r < 8; ++r)
      outB[(size_t)(iTile + r + 8 * lhalf) * DF + 16 * g + lcol] =
          (_Float16)(h[g][r] * rs[r]);
}

// ---------------------------------------------------------------------------
// label[b,n] = sigmoid(X[b,n,:] . Wd + bd)
// ---------------------------------------------------------------------------
__global__ void label_kernel(const _Float16* __restrict__ X, const float* __restrict__ Wd,
                             const float* __restrict__ bd, float* __restrict__ out) {
  int n = blockIdx.x * blockDim.x + threadIdx.x;
  if (n >= BATCH * NN) return;
  const _Float16* x = X + (size_t)n * DF;
  float s = bd[0];
#pragma unroll 8
  for (int d = 0; d < DF; ++d) s += (float)x[d] * Wd[d];
  out[n] = 1.0f / (1.0f + __expf(-s));
}

// ---------------------------------------------------------------------------
extern "C" void kernel_launch(void* const* d_in, const int* in_sizes, int n_in,
                              void* d_out, int out_size, void* d_ws, size_t ws_size,
                              hipStream_t stream) {
  const float* features = (const float*)d_in[0];
  const int*   adj      = (const int*)d_in[1];
  const float* W0 = (const float*)d_in[2];
  const float* b0 = (const float*)d_in[3];
  const float* W1 = (const float*)d_in[4];
  const float* b1 = (const float*)d_in[5];
  const float* Wd = (const float*)d_in[6];
  const float* bd = (const float*)d_in[7];
  float* out = (float*)d_out;

  char* ws = (char*)d_ws;
  const size_t XBYTES = (size_t)BATCH * NN * DF * sizeof(_Float16);   // 4 MB
  const size_t PBYTES = (size_t)BATCH * NN * DF * sizeof(float);      // 8 MB per split
  _Float16* X0     = (_Float16*)(ws);
  _Float16* X1     = (_Float16*)(ws + XBYTES);
  _Float16* AGG    = (_Float16*)(ws + 2 * XBYTES);
  float*    AGGP   = (float*)(ws + 3 * XBYTES);                       // KSPLIT partials
  float*    degbuf = (float*)(ws + 3 * XBYTES + KSPLIT * PBYTES);
  float*    invdeg = (float*)(ws + 3 * XBYTES + KSPLIT * PBYTES
                              + (size_t)BATCH * NN * sizeof(float));

  hipMemsetAsync(degbuf, 0, (size_t)BATCH * NN * sizeof(float), stream);
  dim3 dgrid(NN / 256, NN / 256, BATCH);
  deg_partial_kernel<<<dgrid, 256, 0, stream>>>(adj, degbuf);
  invdeg_kernel<<<(BATCH * NN + 255) / 256, 256, 0, stream>>>(degbuf, invdeg);
  const int n8 = BATCH * NN * DF / 8;
  cvt_kernel<<<(n8 + 255) / 256, 256, 0, stream>>>(features, X0, n8);

  dim3 ag(NN / 128, KSPLIT, BATCH);
  dim3 dg(NN / 128, BATCH);
  const int nred = BATCH * NN * DF / 4;

  agg_kernel   <<<ag, 256, 0, stream>>>(adj, X0, AGGP);
  reduce_kernel<<<(nred + 255) / 256, 256, 0, stream>>>(AGGP, invdeg, AGG);
  dense_kernel <<<dg, 256, 0, stream>>>(X0, AGG, W0, b0, X1);

  agg_kernel   <<<ag, 256, 0, stream>>>(adj, X1, AGGP);
  reduce_kernel<<<(nred + 255) / 256, 256, 0, stream>>>(AGGP, invdeg, AGG);
  dense_kernel <<<dg, 256, 0, stream>>>(X1, AGG, W1, b1, X0);

  label_kernel<<<(BATCH * NN + 255) / 256, 256, 0, stream>>>(X0, Wd, bd, out);
}
